// ClassifierModule_37847251813156
// MI455X (gfx1250) — matrix operand order repaired and checked by reading, not compiled, not run on hardware
//
#include <hip/hip_runtime.h>
#include <hip/hip_bf16.h>

// ---------------- CDNA5 WMMA fragment types ----------------
typedef __attribute__((ext_vector_type(16))) __bf16 v16bf;
typedef __attribute__((ext_vector_type(8)))  float  v8f;
typedef __attribute__((ext_vector_type(2)))  float  v2f;

union BF16x16 { v16bf v; uint4 q[2]; };

__device__ __forceinline__ unsigned short f32_to_bf16_bits(float f) {
  unsigned int u = __float_as_uint(f);
  u += 0x7FFFu + ((u >> 16) & 1u);   // round-to-nearest-even
  return (unsigned short)(u >> 16);
}

// ---------------- problem dims ----------------
#define NB   8
#define CIN  2048
#define HW   40
#define NSP  1600
#define CO   19
#define NA   4096
#define NL   64
#define NK   25
#define OHW  320

// ---------------- workspace layout (bytes) ----------------
#define OFF_XPK    ((size_t)0)          // bf16 [8][1600][2048]     52,428,800 B
#define OFF_APACK  ((size_t)52428800)   // bf16 packed W fragments   4,718,592 B
#define OFF_BSUM   ((size_t)57147392)   // f32  [32]                       128 B
#define OFF_ZERO   ((size_t)57147520)   // bf16 zero page                 4096 B
#define OFF_CONV   ((size_t)57151616)   // f32  [8][19][1600]          972,800 B
#define OFF_FEA    ((size_t)58124416)   // f32  [8][19][1600]          972,800 B
#define OFF_RECIP  ((size_t)59097216)   // f32  [8*19*4096]          2,490,368 B
#define OFF_ATT    ((size_t)61587584)   // f32  [8*19*4096*25]      62,259,200 B

#define OUT0_ELEMS ((size_t)15564800)   // 8*19*320*320

// =====================================================================
// 1) x: NCHW f32 -> NHWC bf16 via LDS tile transpose (coalesced both sides)
// =====================================================================
__global__ __launch_bounds__(256) void k_pack_x(const float* __restrict__ x,
                                                unsigned short* __restrict__ xpk) {
  __shared__ float tile[32][33];
  const int b  = blockIdx.z;
  const int cb = blockIdx.y * 32;   // channel base
  const int nb = blockIdx.x * 32;   // spatial base
  const int tx = threadIdx.x, ty = threadIdx.y;   // (32, 8)
  for (int it = 0; it < 4; ++it) {
    int ci = cb + ty + it * 8;
    tile[ty + it * 8][tx] = x[((size_t)b * CIN + ci) * NSP + nb + tx];
  }
  __syncthreads();
  for (int it = 0; it < 4; ++it) {
    int n = nb + ty + it * 8;
    xpk[((size_t)b * NSP + n) * CIN + cb + tx] = f32_to_bf16_bits(tile[tx][ty + it * 8]);
  }
}

// =====================================================================
// 2) Pack W [4][19][2048][3][3] f32 into bf16 A-fragments for
//    v_wmma_f32_16x16x32_bf16 in contiguous k order (same as B):
//      lane<16 : M=lane,    K = {2v,2v+1} (v<8)
//      lane>=16: M=lane-16, K += 16
//    Rows m>=19 are zero (C_out pad to 32 = two 16-row M tiles).
// =====================================================================
__global__ __launch_bounds__(32) void k_pack_w(const float* __restrict__ W,
                                               unsigned short* __restrict__ apack) {
  const int lane = threadIdx.x;
  const int bx   = blockIdx.x;           // 36*64*2 = 4608
  const int mt   = bx & 1;
  const int kc   = (bx >> 1) & 63;
  const int tap  = bx >> 7;
  const int di = tap / 9, rr = tap % 9;
  const int kh = rr / 3, kw = rr % 3;
  const int m  = (lane & 15) + 16 * mt;
  const int khalf = (lane >> 4) * 16;

  union { unsigned short s[16]; uint4 q[2]; } vals;
  for (int v = 0; v < 8; ++v) {
    int Kv = 2 * v + khalf;
    for (int t = 0; t < 2; ++t) {
      int ci = kc * 32 + Kv + t;
      float f = 0.0f;
      if (m < CO) f = W[(((size_t)di * CO + m) * CIN + ci) * 9 + kh * 3 + kw];
      vals.s[2 * v + t] = f32_to_bf16_bits(f);
    }
  }
  size_t o = ((((size_t)tap * 64 + kc) * 2 + mt) * 32 + lane) * 16;
  uint4* dst = (uint4*)(apack + o);
  dst[0] = vals.q[0];
  dst[1] = vals.q[1];
}

// =====================================================================
// 3) Summed biases over the 4 dilation branches + bf16 zero page used by
//    the conv kernel to realize zero-padding without in-loop EXEC masking.
// =====================================================================
__global__ __launch_bounds__(256) void k_bias_zero(const float* __restrict__ bvec,
                                                   float* __restrict__ bsum,
                                                   unsigned short* __restrict__ zpage) {
  int t = threadIdx.x;
  if (t < CO) bsum[t] = bvec[t] + bvec[CO + t] + bvec[2 * CO + t] + bvec[3 * CO + t];
  for (int i = t; i < 2048; i += 256) zpage[i] = 0;
}

// =====================================================================
// 4) ASPP conv as implicit GEMM with v_wmma_f32_16x16x32_bf16.
//    1 wave per block; each wave owns TWO 16-column spatial tiles so each
//    A (weight) fragment feeds two WMMAs: 8 loads / 4 WMMAs per K-chunk.
//    K loop: 36 taps x 64 chunks of 32 channels.
//    - Halo taps read a zero page (per-lane pointer select hoisted out of
//      the K loop -> no EXEC divergence / hazard NOPs around WMMA).
//    - Two-slot rotation: loads for chunk k+1 issue before the 4 WMMAs of
//      chunk k, so each WMMA group waits only on 4-WMMA-old loads.
// =====================================================================
__device__ __forceinline__ void load_kc(const unsigned short* __restrict__ xb0,
                                        const unsigned short* __restrict__ xb1,
                                        const unsigned short* __restrict__ ab,
                                        int kc, BF16x16& a0, BF16x16& a1,
                                        BF16x16& b0, BF16x16& b1) {
  const uint4* p;
  p = (const uint4*)(ab + (size_t)(kc * 2 + 0) * 512); a0.q[0] = p[0]; a0.q[1] = p[1];
  p = (const uint4*)(ab + (size_t)(kc * 2 + 1) * 512); a1.q[0] = p[0]; a1.q[1] = p[1];
  p = (const uint4*)(xb0 + kc * 32);                   b0.q[0] = p[0]; b0.q[1] = p[1];
  p = (const uint4*)(xb1 + kc * 32);                   b1.q[0] = p[0]; b1.q[1] = p[1];
}

#define WMMA_BF16(ACC, A, Bf) \
  ACC = __builtin_amdgcn_wmma_f32_16x16x32_bf16(false, (A).v, false, (Bf).v, \
                                                (short)0, ACC, false, false)

__global__ __launch_bounds__(32) void k_conv(const unsigned short* __restrict__ xpk,
                                             const unsigned short* __restrict__ apack,
                                             const float* __restrict__ bsum,
                                             const unsigned short* __restrict__ zpage,
                                             float* __restrict__ conv) {
  const int lane = threadIdx.x;
  const int b    = blockIdx.x;    // 8
  const int ntp  = blockIdx.y;    // 50: pair of adjacent 16-col tiles
  const int nn   = lane & 15;
  const int hi   = lane >> 4;
  const int n0   = ntp * 32 + nn;
  const int n1   = n0 + 16;
  const int h0 = n0 / HW, w0 = n0 % HW;
  const int h1 = n1 / HW, w1 = n1 % HW;

  v8f acc00 = {}, acc10 = {};   // tile0: out-ch 0..15 / 16..31
  v8f acc01 = {}, acc11 = {};   // tile1

  for (int tap = 0; tap < 36; ++tap) {
    const int di = tap / 9, rr = tap % 9;
    const int d  = (di == 0) ? 6 : (di == 1) ? 12 : (di == 2) ? 18 : 24;
    const int kh = rr / 3, kw = rr % 3;
    const int dh = d * (kh - 1), dw = d * (kw - 1);

    const int hh0 = h0 + dh, ww0 = w0 + dw;
    const int hh1 = h1 + dh, ww1 = w1 + dw;
    const bool v0 = (hh0 >= 0) && (hh0 < HW) && (ww0 >= 0) && (ww0 < HW);
    const bool v1 = (hh1 >= 0) && (hh1 < HW) && (ww1 >= 0) && (ww1 < HW);
    // Per-lane B source: real row or the zero page (padding halo).
    // lane<16 -> K 0..15, lane>=16 -> K 16..31 of each 32-channel chunk.
    const unsigned short* xb0 =
        v0 ? (xpk + (((size_t)b * HW + hh0) * HW + ww0) * CIN + hi * 16)
           : (zpage + hi * 16);
    const unsigned short* xb1 =
        v1 ? (xpk + (((size_t)b * HW + hh1) * HW + ww1) * CIN + hi * 16)
           : (zpage + hi * 16);
    const unsigned short* ab = apack + (size_t)tap * 64 * 1024 + lane * 16;
    __builtin_prefetch((const void*)xb0, 0, 1);   // global_prefetch_b8
    __builtin_prefetch((const void*)xb1, 0, 1);

    // Two-slot software pipeline (named regs; kc even -> slot A, odd -> slot B)
    BF16x16 a0A, a1A, b0A, b1A, a0B, a1B, b0B, b1B;
    load_kc(xb0, xb1, ab, 0, a0A, a1A, b0A, b1A);
    for (int kc = 0; kc < 64; kc += 2) {
      load_kc(xb0, xb1, ab, kc + 1, a0B, a1B, b0B, b1B);
      WMMA_BF16(acc00, a0A, b0A);
      WMMA_BF16(acc01, a0A, b1A);
      WMMA_BF16(acc10, a1A, b0A);
      WMMA_BF16(acc11, a1A, b1A);
      if (kc + 2 < 64) load_kc(xb0, xb1, ab, kc + 2, a0A, a1A, b0A, b1A);
      WMMA_BF16(acc00, a0B, b0B);
      WMMA_BF16(acc01, a0B, b1B);
      WMMA_BF16(acc10, a1B, b0B);
      WMMA_BF16(acc11, a1B, b1B);
    }
  }
  __builtin_amdgcn_s_wait_tensorcnt(0);
  // C layout: VGPR r, lane -> (M = r + 8*hi, N = lane&15)
  float* cb0 = conv + (size_t)b * CO * NSP + n0;
  float* cb1 = conv + (size_t)b * CO * NSP + n1;
  for (int r = 0; r < 8; ++r) {
    int c = r + 8 * hi;
    float bs = bsum[c];
    cb0[(size_t)c * NSP] = acc00[r] + bs;
    cb1[(size_t)c * NSP] = acc01[r] + bs;
  }
  if (hi == 0) {
    for (int r = 0; r < 3; ++r) {
      int c = 16 + r;
      float bs = bsum[c];
      cb0[(size_t)c * NSP] = acc10[r] + bs;
      cb1[(size_t)c * NSP] = acc11[r] + bs;
    }
  }
}

// =====================================================================
// 5) Fused bilinear-x8 (align_corners) upsample -> argmax(19) -> 8x8 mean
//    pool of the one-hot == per-channel vote count / 64. One block per
//    (b, i, j) pooled cell; conv map (0.97 MB) lives in L2.
// =====================================================================
__device__ __forceinline__ float bilin40(const float* __restrict__ p, int oh, int ow) {
  const float s = 39.0f / 319.0f;
  float fh = oh * s, fw = ow * s;
  int h0 = (int)fh; if (h0 > 38) h0 = 38;
  int w0 = (int)fw; if (w0 > 38) w0 = 38;
  float ah = fh - (float)h0, aw = fw - (float)w0;
  const float* r0 = p + h0 * HW + w0;
  float v00 = r0[0], v01 = r0[1], v10 = r0[HW], v11 = r0[HW + 1];
  return (v00 * (1.0f - aw) + v01 * aw) * (1.0f - ah)
       + (v10 * (1.0f - aw) + v11 * aw) * ah;
}

__global__ __launch_bounds__(64) void k_argmax_pool(const float* __restrict__ conv,
                                                    float* __restrict__ feamap) {
  __shared__ int counts[CO];
  const int tid = threadIdx.x;
  const int j = blockIdx.x, i = blockIdx.y, b = blockIdx.z;
  if (tid < CO) counts[tid] = 0;
  __syncthreads();
  const int oh = i * 8 + (tid >> 3);
  const int ow = j * 8 + (tid & 7);
  const float* cb = conv + (size_t)b * CO * NSP;
  float best = -3.4e38f; int bi = 0;
  for (int c = 0; c < CO; ++c) {
    float v = bilin40(cb + (size_t)c * NSP, oh, ow);
    if (v > best) { best = v; bi = c; }   // first-max tie rule (strict >)
  }
  atomicAdd(&counts[bi], 1);
  __syncthreads();
  if (tid < CO)
    feamap[((size_t)b * CO + tid) * NSP + i * HW + j] = (float)counts[tid] * (1.0f / 64.0f);
}

// =====================================================================
// 6) Fused: passthrough copy of attentions to the 2nd output + per-row
//    nonzero count -> recip = 1/(nnz + 1e-5). 64 lanes per row, 4 rows/blk.
// =====================================================================
__global__ __launch_bounds__(256) void k_nz_copy(const float* __restrict__ attn,
                                                 float* __restrict__ recip,
                                                 float* __restrict__ att_out) {
  __shared__ int cnt[4];
  const int tid = threadIdx.x;
  const size_t rowbase = (size_t)blockIdx.x * 4;
  const int rsub = tid >> 6, l = tid & 63;
  if (tid < 4) cnt[tid] = 0;
  __syncthreads();
  size_t off = (rowbase + rsub) * NL + l;
  float v = attn[off];
  att_out[off] = v;
  if (v != 0.0f) atomicAdd(&cnt[rsub], 1);
  __syncthreads();
  if (l == 0) recip[rowbase + rsub] = 1.0f / ((float)cnt[rsub] + 1e-5f);
}

// =====================================================================
// 7) Attention einsum att[b,c,a,k] = sum_l (A[a,l]*recip[a]) * U[k,l]
//    as a 4096x25x64 GEMM per (b,c) with v_wmma_f32_16x16x4_f32.
//    U (the 5x5 unfold of feamap) staged in LDS, stride 33 -> conflict-free
//    transposed B-fragment gathers. Row scale applied at store.
// =====================================================================
__global__ __launch_bounds__(256) void k_att(const float* __restrict__ attn,
                                             const float* __restrict__ feamap,
                                             const float* __restrict__ recip,
                                             float* __restrict__ att) {
  __shared__ float Ut[64 * 33];           // Ut[l*33 + k] = U[k][l], k>=25 zero
  const int bc   = blockIdx.x;            // 0..151
  const int mgrp = blockIdx.y;            // 0..7
  const int tid  = threadIdx.x;
  const float* fm = feamap + (size_t)bc * NSP;
  for (int idx = tid; idx < 64 * 33; idx += 256) {
    int l = idx / 33, k = idx % 33;
    float v = 0.0f;
    if (k < NK) {
      int kh = k / 5, kw = k % 5, ph = l >> 3, pw = l & 7;
      v = fm[(ph * 5 + kh) * HW + (pw * 5 + kw)];
    }
    Ut[idx] = v;
  }
  __syncthreads();

  const int lane = tid & 31, wv = tid >> 5;
  const int nn = lane & 15, hi = lane >> 4;
  const float* ab = attn + (size_t)bc * NA * NL;
  const float* rb = recip + (size_t)bc * NA;
  float* ob = att + (size_t)bc * NA * NK;

  for (int q = 0; q < 4; ++q) {
    const int mt = mgrp * 32 + wv * 4 + q;          // 0..255
    const int arow = mt * 16 + nn;
    const float* ap = ab + (size_t)arow * NL + 2 * hi;
    for (int ntile = 0; ntile < 2; ++ntile) {
      const int k = ntile * 16 + nn;
      v8f acc = {};
      for (int kk = 0; kk < 16; ++kk) {
        const float2* f2 = (const float2*)(ap + kk * 4);
        v2f a; a.x = f2->x; a.y = f2->y;             // A: K = kk*4 + 2*hi + {0,1}
        int l = kk * 4 + 2 * hi;
        v2f bb; bb.x = Ut[l * 33 + k]; bb.y = Ut[(l + 1) * 33 + k];
        acc = __builtin_amdgcn_wmma_f32_16x16x4_f32(false, a, false, bb,
                                                    (short)0, acc, false, false);
      }
      if (k < NK) {
        float* op = ob + k;
        for (int r = 0; r < 8; ++r) {
          int ai = mt * 16 + r + 8 * hi;
          op[(size_t)ai * NK] = acc[r] * rb[ai];
        }
      }
    }
  }
}

// =====================================================================
// 8) Final: fold att back (no-overlap 5x5), recompute bilinear upsample,
//    out = up * (1 + corr). Coalesced writes of the 62 MB output.
// =====================================================================
__global__ __launch_bounds__(256) void k_final(const float* __restrict__ conv,
                                               const float* __restrict__ att,
                                               float* __restrict__ out0) {
  size_t idx = (size_t)blockIdx.x * 256 + threadIdx.x;
  if (idx >= OUT0_ELEMS) return;
  int ow = (int)(idx % OHW); size_t t = idx / OHW;
  int oh = (int)(t % OHW);   t /= OHW;
  int c  = (int)(t % CO);
  int b  = (int)(t / CO);
  float up = bilin40(conv + ((size_t)b * CO + c) * NSP, oh, ow);
  int a  = (oh / 5) * 64 + (ow / 5);
  int kk = (oh % 5) * 5 + (ow % 5);
  float corr = att[(((size_t)(b * CO + c)) * NA + a) * NK + kk];
  out0[idx] = up * (1.0f + corr);
}

// =====================================================================
extern "C" void kernel_launch(void* const* d_in, const int* in_sizes, int n_in,
                              void* d_out, int out_size, void* d_ws, size_t ws_size,
                              hipStream_t stream) {
  (void)in_sizes; (void)n_in; (void)out_size; (void)ws_size;
  const float* x    = (const float*)d_in[0];   // [8,2048,40,40]
  const float* attn = (const float*)d_in[1];   // [8,19,4096,64]
  const float* W    = (const float*)d_in[2];   // [4,19,2048,3,3]
  const float* bias = (const float*)d_in[3];   // [4,19]
  float* out = (float*)d_out;

  char* ws = (char*)d_ws;
  unsigned short* xpk   = (unsigned short*)(ws + OFF_XPK);
  unsigned short* apack = (unsigned short*)(ws + OFF_APACK);
  float* bsum   = (float*)(ws + OFF_BSUM);
  unsigned short* zpage = (unsigned short*)(ws + OFF_ZERO);
  float* convb  = (float*)(ws + OFF_CONV);
  float* feamap = (float*)(ws + OFF_FEA);
  float* recip  = (float*)(ws + OFF_RECIP);
  float* attw   = (float*)(ws + OFF_ATT);

  k_pack_x<<<dim3(NSP / 32, CIN / 32, NB), dim3(32, 8), 0, stream>>>(x, xpk);
  k_pack_w<<<36 * 64 * 2, 32, 0, stream>>>(W, apack);
  k_bias_zero<<<1, 256, 0, stream>>>(bias, bsum, zpage);
  k_conv<<<dim3(NB, 50), 32, 0, stream>>>(xpk, apack, bsum, zpage, convb);
  k_argmax_pool<<<dim3(HW, HW, NB), 64, 0, stream>>>(convb, feamap);
  k_nz_copy<<<(NB * CO * NA) / 4, 256, 0, stream>>>(attn, recip, out + OUT0_ELEMS);
  k_att<<<dim3(NB * CO, 8), 256, 0, stream>>>(attn, feamap, recip, attw);
  k_final<<<(unsigned)((OUT0_ELEMS + 255) / 256), 256, 0, stream>>>(convb, attw, out);
}
